// LoRALinear_43508018709279
// MI455X (gfx1250) — matrix operand order repaired and checked by reading, not compiled, not run on hardware
//
#include <hip/hip_runtime.h>

// ---------------------------------------------------------------------------
// LoRA Linear for MI455X (gfx1250, wave32, WMMA + async global->LDS)
//   out = x @ W^T + bias + 2.0 * (x @ A^T) @ B^T
// Strategy: fold LoRA into W_eff = W + 2*B@A (bf16), convert x to bf16,
// then one dense bf16 WMMA GEMM (f32 accumulate) with double-buffered LDS
// filled by GLOBAL_LOAD_ASYNC_TO_LDS_B128 (ASYNCcnt), bias epilogue.
// ---------------------------------------------------------------------------

typedef __attribute__((ext_vector_type(16))) __bf16 v16bf;
typedef __attribute__((ext_vector_type(8)))  float  v8f;

#define D_IN    4096
#define D_OUT   4096
#define M_TOT   8192      // B*S = 4*2048
#define R_LORA  16
#define SCALING 2.0f      // 32.0 / 16

#define BM 128
#define BN 128
#define BK 64             // K per LDS stage (2 WMMA substeps of 32)
#define LDT 72            // padded LDS row stride in bf16: 144 B, 16B-aligned
#define KSTAGES (D_IN / BK)

union FragBF { uint4 u[2]; v16bf v; };

// ---- CDNA5 async global->LDS (direct path, no VGPR round trip) -------------
__device__ __forceinline__ void async_ld_b128(const void* gaddr, unsigned lds) {
  asm volatile("global_load_async_to_lds_b128 %0, %1, off"
               :: "v"(lds), "v"(gaddr) : "memory");
}
__device__ __forceinline__ void wait_async0() {
  asm volatile("s_wait_asynccnt 0x0" ::: "memory");
}

// ---- Kernel 1: W_eff[o][d] = bf16( W[o][d] + SCALING * sum_r B[o][r]*A[r][d] )
__global__ __launch_bounds__(256) void build_weff_kernel(
    const float* __restrict__ W, const float* __restrict__ lora_a,
    const float* __restrict__ lora_b, __bf16* __restrict__ Weff) {
  const int o = blockIdx.x;
  float lb[R_LORA];
#pragma unroll
  for (int r = 0; r < R_LORA; ++r) lb[r] = lora_b[o * R_LORA + r] * SCALING;
  for (int d = threadIdx.x; d < D_IN; d += 256) {
    float acc = W[(size_t)o * D_IN + d];
#pragma unroll
    for (int r = 0; r < R_LORA; ++r) acc += lb[r] * lora_a[r * D_IN + d];
    Weff[(size_t)o * D_IN + d] = (__bf16)acc;
  }
}

// ---- Kernel 2: x (f32) -> x (bf16), 4 elements per thread
__global__ __launch_bounds__(256) void cvt_x_kernel(const float* __restrict__ x,
                                                    __bf16* __restrict__ xb) {
  const size_t i = ((size_t)blockIdx.x * 256 + threadIdx.x) * 4;
  const float4 v = *(const float4*)(x + i);
  union { __bf16 h[4]; uint2 u; } p;
  p.h[0] = (__bf16)v.x; p.h[1] = (__bf16)v.y;
  p.h[2] = (__bf16)v.z; p.h[3] = (__bf16)v.w;
  *(uint2*)(xb + i) = p.u;
}

// ---- Kernel 3: out[m][n] = bias[n] + sum_k Xb[m][k] * Weff[n][k]
// 128x128 block tile, BK=64, double-buffered LDS via async copies,
// 8 waves (2x4), wave tile 64x32 = 4x2 WMMA frags, 16 WMMA per stage.
__global__ __launch_bounds__(256) void lora_gemm_kernel(
    const __bf16* __restrict__ Xb, const __bf16* __restrict__ Wb,
    const float* __restrict__ bias, float* __restrict__ out) {
  __shared__ __align__(16) __bf16 As[2][BM * LDT];
  __shared__ __align__(16) __bf16 Bs[2][BN * LDT];

  const int tid   = threadIdx.x;
  const int wave  = tid >> 5;
  const int lane  = tid & 31;
  const int lrow  = lane & 15;      // matrix row/col within fragment
  const int lhalf = lane >> 4;      // K-half selector (contiguous k order used here)
  const int wave_m = wave >> 2;     // 0..1 -> 64-row slab
  const int wave_n = wave & 3;      // 0..3 -> 32-col slab

  const int mBase = blockIdx.y * BM;
  const int nBase = blockIdx.x * BN;

  // Async fill coords: tile rows are 128 B (64 bf16) = 8 x b128 chunks.
  // 256 threads x 4 rounds x 16 B covers one 128x64 tile.
  const int arow = tid >> 3;        // 0..31 (+ l*32)
  const int acol = (tid & 7) * 8;   // bf16 offset, 16 B granules

  const __bf16* gA = Xb + (size_t)(mBase + arow) * D_IN + acol;
  const __bf16* gB = Wb + (size_t)(nBase + arow) * D_IN + acol;
  const unsigned ldsA0 = (unsigned)(size_t)&As[0][arow * LDT + acol];
  const unsigned ldsA1 = (unsigned)(size_t)&As[1][arow * LDT + acol];
  const unsigned ldsB0 = (unsigned)(size_t)&Bs[0][arow * LDT + acol];
  const unsigned ldsB1 = (unsigned)(size_t)&Bs[1][arow * LDT + acol];

  v8f acc[4][2] = {};

  // Prologue: fill buffer 0 with stage 0.
#pragma unroll
  for (int l = 0; l < 4; ++l) {
    async_ld_b128(gA + (size_t)l * 32 * D_IN, ldsA0 + l * 32 * LDT * 2);
    async_ld_b128(gB + (size_t)l * 32 * D_IN, ldsB0 + l * 32 * LDT * 2);
  }

  for (int s = 0; s < KSTAGES; ++s) {
    const int b = s & 1;
    wait_async0();       // my async writes for buffer b have landed
    __syncthreads();     // everyone's writes landed; everyone's reads of
                         // the other buffer are done (dscnt drained here)

    if (s + 1 < KSTAGES) {   // overlap next stage's fill with compute
      const int k = (s + 1) * BK;
      const unsigned lA = b ? ldsA0 : ldsA1;
      const unsigned lB = b ? ldsB0 : ldsB1;
#pragma unroll
      for (int l = 0; l < 4; ++l) {
        async_ld_b128(gA + (size_t)l * 32 * D_IN + k, lA + l * 32 * LDT * 2);
        async_ld_b128(gB + (size_t)l * 32 * D_IN + k, lB + l * 32 * LDT * 2);
      }
    }

    const __bf16* Asb = &As[b][0];
    const __bf16* Bsb = &Bs[b][0];
#pragma unroll
    for (int ks = 0; ks < 2; ++ks) {
      const int kk = ks * 32;
      // B fragment (32x16 bf16): lanes 0-15 hold K=kk..kk+15 of col lrow,
      //                          lanes 16-31 hold K=kk+16..kk+31
      FragBF bf[2];
#pragma unroll
      for (int n = 0; n < 2; ++n) {
        const __bf16* p =
            &Bsb[(wave_n * 32 + n * 16 + lrow) * LDT + kk + lhalf * 16];
        bf[n].u[0] = *(const uint4*)(p);
        bf[n].u[1] = *(const uint4*)(p + 8);
      }
      // A fragment (16x32 bf16): VGPR0-3 K = kk + lhalf*16 .. +7,
      //                          VGPR4-7 K = kk + lhalf*16 + 8 .. +7
      FragBF af[4];
#pragma unroll
      for (int m = 0; m < 4; ++m) {
        const __bf16* p = &Asb[(wave_m * 64 + m * 16 + lrow) * LDT + kk];
        af[m].u[0] = *(const uint4*)(p + lhalf * 16);
        af[m].u[1] = *(const uint4*)(p + lhalf * 16 + 8);
      }
#pragma unroll
      for (int m = 0; m < 4; ++m)
#pragma unroll
        for (int n = 0; n < 2; ++n)
          acc[m][n] = __builtin_amdgcn_wmma_f32_16x16x32_bf16(
              false, af[m].v, false, bf[n].v, (short)0, acc[m][n], false, false);
    }
  }

  // Epilogue: C layout -> lane holds N = lrow, VGPR i holds M = i + 8*lhalf
#pragma unroll
  for (int n = 0; n < 2; ++n) {
    const int col = nBase + wave_n * 32 + n * 16 + lrow;
    const float bv = bias[col];
#pragma unroll
    for (int m = 0; m < 4; ++m) {
      const int rowb = mBase + wave_m * 64 + m * 16 + lhalf * 8;
#pragma unroll
      for (int i = 0; i < 8; ++i)
        out[(size_t)(rowb + i) * D_OUT + col] = acc[m][n][i] + bv;
    }
  }
}

extern "C" void kernel_launch(void* const* d_in, const int* in_sizes, int n_in,
                              void* d_out, int out_size, void* d_ws, size_t ws_size,
                              hipStream_t stream) {
  (void)in_sizes; (void)n_in; (void)out_size; (void)ws_size;

  const float* x    = (const float*)d_in[0];  // [8192, 4096]
  const float* W    = (const float*)d_in[1];  // [4096, 4096]
  const float* bias = (const float*)d_in[2];  // [4096]
  const float* la   = (const float*)d_in[3];  // [16, 4096]
  const float* lb   = (const float*)d_in[4];  // [4096, 16]
  float* out        = (float*)d_out;          // [8192, 4096]

  __bf16* Weff = (__bf16*)d_ws;                                        // 32 MB
  __bf16* Xb   = (__bf16*)((char*)d_ws + (size_t)D_OUT * D_IN * 2);    // 64 MB

  build_weff_kernel<<<D_OUT, 256, 0, stream>>>(W, la, lb, Weff);
  cvt_x_kernel<<<((size_t)M_TOT * D_IN) / (4 * 256), 256, 0, stream>>>(x, Xb);

  dim3 grid(D_OUT / BN, M_TOT / BM);
  lora_gemm_kernel<<<grid, 256, 0, stream>>>(Xb, Weff, bias, out);
}
